// MultiHeadCrossAttentionWithMask_46480136077565
// MI455X (gfx1250) — matrix operand order repaired and checked by reading, not compiled, not run on hardware
//
#include <hip/hip_runtime.h>
#include <hip/hip_bf16.h>

typedef _Float16 half16 __attribute__((ext_vector_type(16)));
typedef _Float16 half8  __attribute__((ext_vector_type(8)));
typedef float    floatx8 __attribute__((ext_vector_type(8)));

constexpr int kB = 2, kS = 2048, kD = 1024, kH = 16, kHD = 64;
constexpr int kM = kB * kS;  // 4096 rows

#define WMMA_F16(a, b, c) \
    __builtin_amdgcn_wmma_f32_16x16x32_f16(false, (a), false, (b), (short)0, (c), false, false)

// A-matrix fragment (16x32 f16): lane L holds row (L&15); element i <-> k = (L>>4)*16 + i
__device__ __forceinline__ half16 ldsA(const _Float16* p, int lane) {
    const int o = (lane >> 4) * 16;
    union { half16 v; half8 h[2]; } u;
    u.h[0] = *(const half8*)(p + o);
    u.h[1] = *(const half8*)(p + o + 8);
    return u.v;
}
// B-matrix fragment (32x16 f16), from Bt[n][k] row-major: lane L holds col (L&15);
// element i <-> k = (L>>4)*16 + i  (contiguous 16 halves)
__device__ __forceinline__ half16 ldsB(const _Float16* p, int lane) {
    const int o = (lane >> 4) * 16;
    union { half16 v; half8 h[2]; } u;
    u.h[0] = *(const half8*)(p + o);
    u.h[1] = *(const half8*)(p + o + 8);
    return u.v;
}

// ---------------------------------------------------------------- weights f32 -> f16
__global__ void wconv_kernel(const float* __restrict__ Wq, const float* __restrict__ Wk,
                             const float* __restrict__ Wv, const float* __restrict__ Wo,
                             _Float16* __restrict__ W16) {
    const int n = kD * kD;
    for (int i = blockIdx.x * blockDim.x + threadIdx.x; i < 4 * n; i += gridDim.x * blockDim.x) {
        const float* src = (i < n) ? Wq : (i < 2 * n) ? Wk : (i < 3 * n) ? Wv : Wo;
        W16[i] = (_Float16)src[i & (n - 1)];
    }
}

// ---------------------------------------------------------------- row LayerNorm -> f16
__global__ __launch_bounds__(256) void ln_kernel(const float* __restrict__ x,
                                                 const float* __restrict__ gamma,
                                                 _Float16* __restrict__ out) {
    const int row = blockIdx.x;
    const int t = threadIdx.x;
    const float4 v = ((const float4*)(x + (size_t)row * kD))[t];
    float s1 = v.x + v.y + v.z + v.w;
    float s2 = v.x * v.x + v.y * v.y + v.z * v.z + v.w * v.w;
#pragma unroll
    for (int off = 16; off > 0; off >>= 1) {
        s1 += __shfl_xor(s1, off, 32);
        s2 += __shfl_xor(s2, off, 32);
    }
    __shared__ float r1[8], r2[8];
    if ((t & 31) == 0) { r1[t >> 5] = s1; r2[t >> 5] = s2; }
    __syncthreads();
    float S1 = 0.f, S2 = 0.f;
#pragma unroll
    for (int i = 0; i < 8; ++i) { S1 += r1[i]; S2 += r2[i]; }
    const float mu   = S1 * (1.0f / kD);
    const float var  = S2 * (1.0f / kD) - mu * mu;
    const float rstd = 1.0f / sqrtf(var + 1e-5f);
    const float4 g = ((const float4*)gamma)[t];
    _Float16* o = out + (size_t)row * kD + t * 4;
    o[0] = (_Float16)((v.x - mu) * rstd * g.x);
    o[1] = (_Float16)((v.y - mu) * rstd * g.y);
    o[2] = (_Float16)((v.z - mu) * rstd * g.z);
    o[3] = (_Float16)((v.w - mu) * rstd * g.w);
}

// ---------------------------------------------------------------- GEMM: out[M,N] = A[M,K] * Bt[N,K]^T + bias
// M = 4096, N = K = 1024. Block tile 128x128, K-step 32, 8 waves (4x2), each wave 2x4 WMMA tiles.
// F32OUT selects the output type at compile time -> branch-free epilogue.
template <bool F32OUT>
__global__ __launch_bounds__(256) void gemm_bt_kernel(
        const _Float16* __restrict__ A, const _Float16* __restrict__ Bt,
        const float* __restrict__ bias, void* __restrict__ outp) {
    constexpr int LT = 40;  // LDS row stride (halves), keeps 16B alignment, breaks bank conflicts
    __shared__ __attribute__((aligned(16))) _Float16 As[128 * LT];
    __shared__ __attribute__((aligned(16))) _Float16 Bs[128 * LT];
    const int t = threadIdx.x;
    const int lane = t & 31, w = t >> 5;
    const int wm = w >> 1, wn = w & 1;
    const int ln = lane & 15, hi = lane >> 4;
    const int r0 = blockIdx.y * 128, c0 = blockIdx.x * 128;
    floatx8 acc[2][4] = {};
    const int rr = t >> 1, sg = t & 1;
    for (int k0 = 0; k0 < kD; k0 += 32) {
        __syncthreads();
        const half8* ga = (const half8*)(A  + (size_t)(r0 + rr) * kD + k0 + sg * 16);
        const half8* gb = (const half8*)(Bt + (size_t)(c0 + rr) * kD + k0 + sg * 16);
        half8 a0 = ga[0], a1 = ga[1];
        half8 b0 = gb[0], b1 = gb[1];
        *(half8*)(As + rr * LT + sg * 16    ) = a0;
        *(half8*)(As + rr * LT + sg * 16 + 8) = a1;
        *(half8*)(Bs + rr * LT + sg * 16    ) = b0;
        *(half8*)(Bs + rr * LT + sg * 16 + 8) = b1;
        __syncthreads();
        half16 af[2], bf[4];
#pragma unroll
        for (int tm = 0; tm < 2; ++tm)
            af[tm] = ldsA(As + (wm * 32 + tm * 16 + ln) * LT, lane);
#pragma unroll
        for (int tn = 0; tn < 4; ++tn)
            bf[tn] = ldsB(Bs + (wn * 64 + tn * 16 + ln) * LT, lane);
#pragma unroll
        for (int tm = 0; tm < 2; ++tm)
#pragma unroll
            for (int tn = 0; tn < 4; ++tn)
                acc[tm][tn] = WMMA_F16(af[tm], bf[tn], acc[tm][tn]);
    }
    // C/D layout: col = lane&15 (+tile), row = v + 8*(lane>>4) (+tile)
#pragma unroll
    for (int tn = 0; tn < 4; ++tn) {
        const int col = c0 + wn * 64 + tn * 16 + ln;
        const float bb = bias[col];
#pragma unroll
        for (int tm = 0; tm < 2; ++tm) {
            const int rbase = r0 + wm * 32 + tm * 16 + 8 * hi;
#pragma unroll
            for (int v = 0; v < 8; ++v) {
                const size_t idx = (size_t)(rbase + v) * kD + col;
                const float val = acc[tm][tn][v] + bb;
                if constexpr (F32OUT) ((float*)outp)[idx] = val;
                else                  ((_Float16*)outp)[idx] = (_Float16)val;
            }
        }
    }
}

// ---------------------------------------------------------------- fused attention
// Block: 128 q-rows of one (b,h). 8 waves; wave w owns q-rows [w*16, w*16+16).
// Pass 1: streaming row max/sum over masked QK^T. Pass 2: recompute scores,
// write normalized probs (f32, to d_out), WMMA-accumulate context with V.
__global__ __launch_bounds__(256) void attn_kernel(
        const _Float16* __restrict__ Q16, const _Float16* __restrict__ K16,
        const _Float16* __restrict__ V16, const int* __restrict__ mask,
        float* __restrict__ probs, _Float16* __restrict__ ctx16) {
    constexpr int LQ = 72;
    __shared__ __attribute__((aligned(16))) _Float16 Qs[128 * LQ];
    __shared__ __attribute__((aligned(16))) _Float16 Ks[64 * LQ];
    __shared__ __attribute__((aligned(16))) _Float16 Vt[64 * LQ];     // Vt[d][j]
    __shared__ __attribute__((aligned(16))) _Float16 Ps[8][16 * LQ];  // per-wave p staging
    __shared__ int Msk[64];
    const int t = threadIdx.x, lane = t & 31, w = t >> 5;
    const int ln = lane & 15, hi = lane >> 4;
    const int q0 = blockIdx.x * 128;
    const int bh = blockIdx.y, b = bh >> 4, h = bh & 15;
    const size_t rowbase = (size_t)b * kS;
    const float NEGINF = -__builtin_inff();

    {   // Q tile 128x64
        const int r = t >> 1, sg = t & 1;
        const half8* gq = (const half8*)(Q16 + (rowbase + q0 + r) * kD + h * kHD + sg * 32);
        half8 a = gq[0], c = gq[1], d = gq[2], e = gq[3];
        *(half8*)(Qs + r * LQ + sg * 32     ) = a;
        *(half8*)(Qs + r * LQ + sg * 32 +  8) = c;
        *(half8*)(Qs + r * LQ + sg * 32 + 16) = d;
        *(half8*)(Qs + r * LQ + sg * 32 + 24) = e;
    }
    float m[8], l[8];
#pragma unroll
    for (int v = 0; v < 8; ++v) { m[v] = NEGINF; l[v] = 0.0f; }
    const int kr = t >> 2, ksg = t & 3;

    // -------- pass 1: row max / row sum
    for (int j0 = 0; j0 < kS; j0 += 64) {
        __syncthreads();
        {
            const half8* gk = (const half8*)(K16 + (rowbase + j0 + kr) * kD + h * kHD + ksg * 16);
            half8 k0v = gk[0], k1v = gk[1];
            *(half8*)(Ks + kr * LQ + ksg * 16    ) = k0v;
            *(half8*)(Ks + kr * LQ + ksg * 16 + 8) = k1v;
            if (t < 64) Msk[t] = mask[(size_t)b * kS + j0 + t];
        }
        __syncthreads();
        floatx8 S[4] = {};
#pragma unroll
        for (int c = 0; c < 2; ++c) {
            half16 aq = ldsA(Qs + (w * 16 + ln) * LQ + c * 32, lane);
#pragma unroll
            for (int tt = 0; tt < 4; ++tt) {
                half16 bk = ldsB(Ks + (tt * 16 + ln) * LQ + c * 32, lane);
                S[tt] = WMMA_F16(aq, bk, S[tt]);
            }
        }
        float tmax[8];
#pragma unroll
        for (int v = 0; v < 8; ++v) tmax[v] = NEGINF;
#pragma unroll
        for (int tt = 0; tt < 4; ++tt) {
            const bool dead = (Msk[tt * 16 + ln] == 0);
#pragma unroll
            for (int v = 0; v < 8; ++v) {
                float s = dead ? NEGINF : S[tt][v];
                S[tt][v] = s;
                tmax[v] = fmaxf(tmax[v], s);
            }
        }
#pragma unroll
        for (int v = 0; v < 8; ++v) {
#pragma unroll
            for (int off = 1; off < 16; off <<= 1)
                tmax[v] = fmaxf(tmax[v], __shfl_xor(tmax[v], off, 16));
        }
#pragma unroll
        for (int v = 0; v < 8; ++v) {
            const float mn = fmaxf(m[v], tmax[v]);
            const float me = fmaxf(mn, -1.0e38f);  // avoid inf-inf
            float ps = 0.0f;
#pragma unroll
            for (int tt = 0; tt < 4; ++tt) ps += __expf(S[tt][v] - me);
#pragma unroll
            for (int off = 1; off < 16; off <<= 1) ps += __shfl_xor(ps, off, 16);
            l[v] = l[v] * __expf(m[v] - me) + ps;
            m[v] = mn;
        }
    }
    float meff[8], rl[8];
#pragma unroll
    for (int v = 0; v < 8; ++v) { meff[v] = fmaxf(m[v], -1.0e38f); rl[v] = 1.0f / l[v]; }
    floatx8 ctx[4] = {};

    // -------- pass 2: probs out + context accumulate
    for (int j0 = 0; j0 < kS; j0 += 64) {
        __syncthreads();
        {
            const half8* gk = (const half8*)(K16 + (rowbase + j0 + kr) * kD + h * kHD + ksg * 16);
            half8 k0v = gk[0], k1v = gk[1];
            *(half8*)(Ks + kr * LQ + ksg * 16    ) = k0v;
            *(half8*)(Ks + kr * LQ + ksg * 16 + 8) = k1v;
            const _Float16* gv = V16 + (rowbase + j0 + kr) * kD + h * kHD + ksg * 16;
            half8 v0 = *(const half8*)gv;
            half8 v1 = *(const half8*)(gv + 8);
#pragma unroll
            for (int i = 0; i < 8; ++i) Vt[(ksg * 16 + i) * LQ + kr] = v0[i];
#pragma unroll
            for (int i = 0; i < 8; ++i) Vt[(ksg * 16 + 8 + i) * LQ + kr] = v1[i];
            if (t < 64) Msk[t] = mask[(size_t)b * kS + j0 + t];
        }
        __syncthreads();
        floatx8 S[4] = {};
#pragma unroll
        for (int c = 0; c < 2; ++c) {
            half16 aq = ldsA(Qs + (w * 16 + ln) * LQ + c * 32, lane);
#pragma unroll
            for (int tt = 0; tt < 4; ++tt) {
                half16 bk = ldsB(Ks + (tt * 16 + ln) * LQ + c * 32, lane);
                S[tt] = WMMA_F16(aq, bk, S[tt]);
            }
        }
#pragma unroll
        for (int tt = 0; tt < 4; ++tt) {
            const bool dead = (Msk[tt * 16 + ln] == 0);
            const int col = j0 + tt * 16 + ln;
#pragma unroll
            for (int v = 0; v < 8; ++v) {
                const float s = dead ? NEGINF : S[tt][v];
                const float p = __expf(s - meff[v]) * rl[v];
                const int qloc = v + 8 * hi;
                probs[((size_t)bh * kS + (q0 + w * 16 + qloc)) * kS + col] = p;
                Ps[w][qloc * LQ + tt * 16 + ln] = (_Float16)p;
            }
        }
        // same-wave LDS ops are in-order -> staging store/load needs no barrier
#pragma unroll
        for (int c = 0; c < 2; ++c) {
            half16 ap = ldsA(&Ps[w][ln * LQ] + c * 32, lane);
#pragma unroll
            for (int td = 0; td < 4; ++td) {
                half16 bv = ldsB(Vt + (td * 16 + ln) * LQ + c * 32, lane);
                ctx[td] = WMMA_F16(ap, bv, ctx[td]);
            }
        }
    }
#pragma unroll
    for (int td = 0; td < 4; ++td) {
        const int col = h * kHD + td * 16 + ln;
#pragma unroll
        for (int v = 0; v < 8; ++v) {
            const int row = q0 + w * 16 + v + 8 * hi;
            ctx16[(rowbase + row) * kD + col] = (_Float16)ctx[td][v];
        }
    }
}

// ----------------------------------------------------------------
extern "C" void kernel_launch(void* const* d_in, const int* in_sizes, int n_in,
                              void* d_out, int out_size, void* d_ws, size_t ws_size,
                              hipStream_t stream) {
    (void)in_sizes; (void)n_in; (void)out_size; (void)ws_size;
    const float* q     = (const float*)d_in[0];
    const float* k     = (const float*)d_in[1];
    const float* v     = (const float*)d_in[2];
    const int*   mask  = (const int*)d_in[3];
    const float* gamma = (const float*)d_in[4];
    const float* Wq = (const float*)d_in[5];  const float* bq = (const float*)d_in[6];
    const float* Wk = (const float*)d_in[7];  const float* bk = (const float*)d_in[8];
    const float* Wv = (const float*)d_in[9];  const float* bv = (const float*)d_in[10];
    const float* Wo = (const float*)d_in[11]; const float* bo = (const float*)d_in[12];

    float* out   = (float*)d_out;                    // context: [B,S,D] f32
    float* probs = out + (size_t)kM * kD;            // probs:   [B,H,S,S] f32

    _Float16* ws  = (_Float16*)d_ws;
    _Float16* W16 = ws;                              // 4 * D*D halves
    _Float16* X16 = W16 + (size_t)4 * kD * kD;       // M*D (reused: xhat, then ctx)
    _Float16* Q16 = X16 + (size_t)kM * kD;
    _Float16* K16 = Q16 + (size_t)kM * kD;
    _Float16* V16 = K16 + (size_t)kM * kD;
    _Float16* C16 = X16;                             // reuse xhat buffer for context

    wconv_kernel<<<2048, 256, 0, stream>>>(Wq, Wk, Wv, Wo, W16);

    dim3 gGrid(kD / 128, kM / 128);
    ln_kernel<<<kM, 256, 0, stream>>>(q, gamma, X16);
    gemm_bt_kernel<false><<<gGrid, 256, 0, stream>>>(X16, W16 + (size_t)0 * kD * kD, bq, Q16);
    ln_kernel<<<kM, 256, 0, stream>>>(k, gamma, X16);
    gemm_bt_kernel<false><<<gGrid, 256, 0, stream>>>(X16, W16 + (size_t)1 * kD * kD, bk, K16);
    ln_kernel<<<kM, 256, 0, stream>>>(v, gamma, X16);
    gemm_bt_kernel<false><<<gGrid, 256, 0, stream>>>(X16, W16 + (size_t)2 * kD * kD, bv, V16);

    attn_kernel<<<dim3(kS / 128, kB * kH), 256, 0, stream>>>(Q16, K16, V16, mask, probs, C16);

    gemm_bt_kernel<true><<<gGrid, 256, 0, stream>>>(C16, W16 + (size_t)3 * kD * kD, bo, out);
}
